// ProRandConvNet_17609365914086
// MI455X (gfx1250) — matrix operand order repaired and checked by reading, not compiled, not run on hardware
//
#include <hip/hip_runtime.h>
#include <math.h>
#include <string.h>

// ---------------------------------------------------------------------------
// ProRandConvNet on MI455X (gfx1250): 10x [deform_conv2d -> instance_norm -> tanh]
//
// deform conv einsum (K=27 contraction, 3 out channels) mapped to
// v_wmma_f32_16x16x32_f16:  D[16x16 f32] = A[16x32 f16 weights] x B[32x16 f16 samples]
//   M = output channel (rows 0..2 valid, rest zero-padded)
//   N = pixel within a 16-pixel tile (one pixel per lane 0..15)
//   K = c*9 + tap (0..26 valid, 27..31 zero-padded)
// ---------------------------------------------------------------------------

typedef __attribute__((ext_vector_type(16))) _Float16 v16h;
typedef __attribute__((ext_vector_type(8)))  float    v8f;

#define Bn   32
#define Cn   3
#define Hn   224
#define Wn   224
#define HWn  (Hn * Wn)                 // 50176
#define KTAP 9
#define KDIM 27                        // Cn * KTAP
#define NTOT ((size_t)Bn * Cn * HWn)   // 4,816,896 elements

#define TILES_PER_IMG (HWn / 16)       // 3136 (exact)
#define NTILES (Bn * TILES_PER_IMG)    // 100352
#define WAVES_PER_BLOCK 8
#define CONV_BLOCKS (NTILES / WAVES_PER_BLOCK)  // 12544 (exact, so all b uniform per block)
#define NORM_BLOCKS ((int)(NTOT / 256))         // 18816 (exact)

#define EPS_IN 1e-5f
#define LNUM   10

// ---------------------------------------------------------------------------
// Build the A-matrix (weights) in WMMA 16-bit A-layout, once per launch.
// 16-bit A 16x32 layout (wave32):
//   lane L (<16):  VGPR v<4 -> K={2v,2v+1};      v>=4 -> K={16+2(v-4), ...+1}   (row M=L)
//   lane L+16:     VGPR v<4 -> K={8+2v, 9+2v};   v>=4 -> K={24+2(v-4), ...+1}   (row M=L)
// Table is [32 lanes][8 dwords], each dword = packed (f16 lo = even K, hi = odd K).
// weight tensor is [o][c][ky][kx] -> flat W[o][k], k = c*9 + ky*3 + kx.
// ---------------------------------------------------------------------------
__global__ void build_wtab_kernel(const float* __restrict__ w, unsigned* __restrict__ wtab) {
  const int lane = threadIdx.x;
  if (lane >= 32) return;
  const int M = lane & 15;
  #pragma unroll
  for (int v = 0; v < 8; ++v) {
    int kbase;
    if (v < 4) kbase = (lane < 16) ? (2 * v)            : (8 + 2 * v);
    else       kbase = (lane < 16) ? (16 + 2 * (v - 4)) : (24 + 2 * (v - 4));
    float f0 = 0.f, f1 = 0.f;
    if (M < Cn) {
      if (kbase     < KDIM) f0 = w[M * KDIM + kbase];
      if (kbase + 1 < KDIM) f1 = w[M * KDIM + kbase + 1];
    }
    union { _Float16 h[2]; unsigned u; } p;
    p.h[0] = (_Float16)f0;
    p.h[1] = (_Float16)f1;
    wtab[lane * 8 + v] = p.u;
  }
}

__global__ void zero_stats_kernel(float* __restrict__ stats) {
  if (threadIdx.x < 2 * Bn * Cn) stats[threadIdx.x] = 0.f;
}

// ---------------------------------------------------------------------------
// Deformable conv via WMMA. One wave per 16-pixel tile. Also accumulates
// per-(b,c) sum / sum-of-squares for the following instance norm.
// ---------------------------------------------------------------------------
__global__ __launch_bounds__(256) void deform_conv_wmma_kernel(
    const float* __restrict__ x,       // [B][C][H][W] activations (iter input)
    const float* __restrict__ offset,  // [B][18][H][W]
    const unsigned* __restrict__ wtab, // [32][8] packed f16 A-matrix
    float* __restrict__ out,           // [B][C][H][W] conv result
    float* __restrict__ stats)         // [B][C][2] {sum, sumsq}
{
  __shared__ float s_acc[6];           // sum[3], sumsq[3] per block
  if (threadIdx.x < 6) s_acc[threadIdx.x] = 0.f;
  __syncthreads();

  const int wave = threadIdx.x >> 5;
  const int lane = threadIdx.x & 31;
  const int tile = blockIdx.x * WAVES_PER_BLOCK + wave;      // grid exact, no guard
  const int b    = tile / TILES_PER_IMG;
  const int pix  = (tile % TILES_PER_IMG) * 16 + (lane & 15);
  const int h    = pix / Wn;
  const int w    = pix - h * Wn;
  const int klo  = (lane < 16) ? 0 : 16;

  const float* __restrict__ offs = offset + (size_t)b * (2 * KTAP) * HWn;
  const float* __restrict__ ximg = x      + (size_t)b * Cn * HWn;

  // Prefetch next tile's offset row into cache (global_prefetch_b8).
  __builtin_prefetch(offs + pix + 16, 0, 1);

  // Per-pixel tap geometry, computed once, shared across the 3 input channels.
  float t_wy[KTAP], t_wx[KTAP];
  int   t_y0[KTAP], t_x0[KTAP];
  #pragma unroll
  for (int tap = 0; tap < KTAP; ++tap) {
    const int ky = tap / 3, kx = tap - 3 * ky;
    const float dy = offs[(2 * tap    ) * HWn + pix];
    const float dx = offs[(2 * tap + 1) * HWn + pix];
    const float py = (float)(h + ky - 1) + dy;
    const float px = (float)(w + kx - 1) + dx;
    const float y0f = floorf(py), x0f = floorf(px);
    t_wy[tap] = py - y0f;  t_wx[tap] = px - x0f;
    t_y0[tap] = (int)y0f;  t_x0[tap] = (int)x0f;
  }

  // B-matrix: this lane's 16 K-slots (K = klo/2 + j + (j & 8)), bilinear-gathered samples.
  v16h bvec;
  #pragma unroll
  for (int j = 0; j < 16; ++j) {
    const int k = (klo >> 1) + j + (j & 8);
    float val = 0.f;
    if (k < KDIM) {
      const int c   = k / KTAP;
      const int tap = k - KTAP * c;
      const float wy = t_wy[tap], wx = t_wx[tap];
      const int   y0 = t_y0[tap], x0 = t_x0[tap];
      const float* __restrict__ plane = ximg + (size_t)c * HWn;
      float acc = 0.f;
      #pragma unroll
      for (int dyc = 0; dyc < 2; ++dyc) {
        const int   yy  = y0 + dyc;
        const float wyc = dyc ? wy : 1.f - wy;
        const int   ycl = min(max(yy, 0), Hn - 1);
        const bool  vy  = (yy >= 0) & (yy < Hn);
        #pragma unroll
        for (int dxc = 0; dxc < 2; ++dxc) {
          const int   xx  = x0 + dxc;
          const float wxc = dxc ? wx : 1.f - wx;
          const int   xcl = min(max(xx, 0), Wn - 1);
          const bool  vx  = (xx >= 0) & (xx < Wn);
          const float wgt = wyc * wxc * ((vy & vx) ? 1.f : 0.f);
          acc += wgt * plane[ycl * Wn + xcl];     // zero-padding semantics
        }
      }
      val = acc;
    }
    bvec[j] = (_Float16)val;
  }

  // A-matrix: preformatted weights (identical for every wave).
  union { unsigned u[8]; v16h h; } avec;
  #pragma unroll
  for (int v = 0; v < 8; ++v) avec.u[v] = wtab[lane * 8 + v];

  // D = A x B + 0   (EXEC is all-ones here: no early returns, divergence reconverged)
  v8f acc8 = {};
  acc8 = __builtin_amdgcn_wmma_f32_16x16x32_f16(
      /*neg_a=*/false, avec.h, /*neg_b=*/false, bvec,
      /*c_mod=*/(short)0, acc8, /*reuse_a=*/false, /*reuse_b=*/false);

  // Lane n (<16) holds out[channel r][pixel n] in acc8[r]; lanes 16..31 hold the
  // zero-padded weight rows (exact 0.0f), so full-wave reductions stay correct.
  float o0 = acc8[0], o1 = acc8[1], o2 = acc8[2];
  if (lane < 16) {
    out[((size_t)b * Cn + 0) * HWn + pix] = o0;
    out[((size_t)b * Cn + 1) * HWn + pix] = o1;
    out[((size_t)b * Cn + 2) * HWn + pix] = o2;
  }

  // Wave-level butterfly reduction of sum / sumsq (lanes >=16 add zeros).
  float s0 = o0, s1 = o1, s2 = o2;
  float q0 = o0 * o0, q1 = o1 * o1, q2 = o2 * o2;
  #pragma unroll
  for (int m = 16; m >= 1; m >>= 1) {
    s0 += __shfl_xor(s0, m, 32);  s1 += __shfl_xor(s1, m, 32);  s2 += __shfl_xor(s2, m, 32);
    q0 += __shfl_xor(q0, m, 32);  q1 += __shfl_xor(q1, m, 32);  q2 += __shfl_xor(q2, m, 32);
  }
  if (lane == 0) {
    atomicAdd(&s_acc[0], s0); atomicAdd(&s_acc[1], s1); atomicAdd(&s_acc[2], s2);
    atomicAdd(&s_acc[3], q0); atomicAdd(&s_acc[4], q1); atomicAdd(&s_acc[5], q2);
  }
  __syncthreads();
  if (threadIdx.x < 6) {
    const int ch    = (threadIdx.x < 3) ? threadIdx.x : threadIdx.x - 3;
    const int which = (threadIdx.x < 3) ? 0 : 1;
    atomicAdd(&stats[((size_t)b * Cn + ch) * 2 + which], s_acc[threadIdx.x]);
  }
}

// ---------------------------------------------------------------------------
// Instance norm (biased var) + tanh.
// ---------------------------------------------------------------------------
__global__ __launch_bounds__(256) void norm_tanh_kernel(
    const float* __restrict__ in, const float* __restrict__ stats,
    const float* __restrict__ gamma, const float* __restrict__ beta,
    float* __restrict__ out)
{
  const size_t i  = (size_t)blockIdx.x * 256 + threadIdx.x;  // grid exact
  const size_t bc = i / HWn;
  const int    ch = (int)(bc % Cn);
  const float  invN = 1.f / (float)HWn;
  const float  mean = stats[bc * 2 + 0] * invN;
  const float  var  = stats[bc * 2 + 1] * invN - mean * mean;
  const float  inv  = rsqrtf(var + EPS_IN);
  const float  y    = (in[i] - mean) * inv * gamma[ch] + beta[ch];
  out[i] = tanhf(y);
}

// ---------------------------------------------------------------------------
// Host-side launch sequence (graph-capture safe: kernels only, fixed count).
// ---------------------------------------------------------------------------
extern "C" void kernel_launch(void* const* d_in, const int* in_sizes, int n_in,
                              void* d_out, int out_size, void* d_ws, size_t ws_size,
                              hipStream_t stream) {
  (void)in_sizes; (void)n_in; (void)out_size; (void)ws_size;

  const float* d_x      = (const float*)d_in[0];
  const float* d_weight = (const float*)d_in[1];
  const float* d_offset = (const float*)d_in[2];
  const float* d_gamma  = (const float*)d_in[3];
  const float* d_beta   = (const float*)d_in[4];
  // d_in[5] = l_num (device int); launch count must be deterministic -> LNUM.

  float* wsf   = (float*)d_ws;
  float* tmp   = wsf;                    // conv result        (NTOT floats)
  float* cur   = wsf + NTOT;             // activation buffer  (NTOT floats)
  float* stats = wsf + 2 * NTOT;         // [B][C][2] sums     (256 floats, padded)
  unsigned* wtab = (unsigned*)(stats + 256);  // [32][8] packed f16 weights

  build_wtab_kernel<<<1, 32, 0, stream>>>(d_weight, wtab);

  for (int it = 0; it < LNUM; ++it) {
    const float* src = (it == 0) ? d_x : cur;
    float*       dst = (it == LNUM - 1) ? (float*)d_out : cur;

    zero_stats_kernel<<<1, 256, 0, stream>>>(stats);
    deform_conv_wmma_kernel<<<CONV_BLOCKS, 256, 0, stream>>>(
        src, d_offset, wtab, tmp, stats);
    norm_tanh_kernel<<<NORM_BLOCKS, 256, 0, stream>>>(
        tmp, stats, d_gamma, d_beta, dst);
  }
}
